// TargetAwareLatentPooler_7748121002259
// MI455X (gfx1250) — compile-verified, matrix operand order checked by reading, not run on hardware
//
#include <hip/hip_runtime.h>

// ---------------------------------------------------------------------------
// TargetAwareLatentPooler for MI455X (gfx1250, wave32, WMMA).
// B=16, N=4096, D=512, L=64. All GEMMs via v_wmma_f32_16x16x32_bf16.
// Global->LDS staging of pre-converted bf16 tiles uses the CDNA5 async path
// (GLOBAL_LOAD_ASYNC_TO_LDS_B128 + s_wait_asynccnt) when available.
// ---------------------------------------------------------------------------

#define B_ 16
#define N_ 4096
#define D_ 512
#define L_ 64

#ifndef __has_builtin
#define __has_builtin(x) 0
#endif

#if __has_builtin(__builtin_amdgcn_global_load_async_to_lds_b128) && \
    __has_builtin(__builtin_amdgcn_s_wait_asynccnt)
#define USE_ASYNC 1
#else
#define USE_ASYNC 0
#endif

typedef __attribute__((ext_vector_type(16))) __bf16 v16bf;
typedef __attribute__((ext_vector_type(8)))  float  v8f;

union AFrag {
    v16bf bf;
    unsigned int u[8];
};

__device__ __forceinline__ unsigned short f2bf(float f) {
    union { float f; unsigned int u; } v; v.f = f;
    unsigned int u = v.u;
    u += 0x7FFFu + ((u >> 16) & 1u);       // round-to-nearest-even
    return (unsigned short)(u >> 16);
}

#if USE_ASYNC
// 16-byte async global->LDS copy (ASYNCcnt-tracked, no VGPR round trip).
// Builtin signature (from clang diagnostic): param0 = __device__ (AS1) int4*,
// param1 = __shared__ (AS3) int4*, then imm offset + cpol.
typedef int v4i __attribute__((vector_size(16)));
typedef __attribute__((address_space(1))) v4i* gptr4;
typedef __attribute__((address_space(3))) v4i* lptr4;

__device__ __forceinline__ void async_copy16(const unsigned short* g, unsigned short* l) {
    v4i* gg = (v4i*)g;    // strips const, generic pointer
    v4i* ll = (v4i*)l;
    __builtin_amdgcn_global_load_async_to_lds_b128((gptr4)gg, (lptr4)ll, 0, 0);
}
__device__ __forceinline__ void async_wait() {
    __builtin_amdgcn_s_wait_asynccnt(0);
}
#endif

// Load a 16x32 bf16 operand fragment from an LDS tile (row stride LD shorts).
// Implements ISA 7.12.2 "16-bit A-Matrix 16x32" layout:
//   lanes 0-15 : rows 0-15, K base 0 ; lanes 16-31 : rows 0-15, K base 8
//   u[0..3] -> K = base + {0,2,4,6} (+1) ; u[4..7] -> K = 16 + base + {0,2,4,6}
// B operand uses the same layout applied to B^T (we always stage B K-contiguous).
template<int LD>
__device__ __forceinline__ AFrag load_frag(const unsigned short* base, int lane) {
    AFrag f;
    const unsigned short* p = base + (lane & 15) * LD + ((lane >> 4) << 3);
#pragma unroll
    for (int v = 0; v < 8; ++v) {
        int k0 = ((v & 4) << 2) + ((v & 3) << 1);
        f.u[v] = *(const unsigned int*)(p + k0);
    }
    return f;
}

__device__ __forceinline__ v8f wmma_bf16(const AFrag& a, const AFrag& b, v8f c) {
    return __builtin_amdgcn_wmma_f32_16x16x32_bf16(
        false, a.bf, false, b.bf, (short)0, c, false, false);
}

__device__ __forceinline__ v8f zero8() {
    v8f z = {0.f, 0.f, 0.f, 0.f, 0.f, 0.f, 0.f, 0.f};
    return z;
}

// ---------------------------------------------------------------------------
// Kernel 0: Wk/Wv fp32 [k][n] -> bf16 transposed [n][k] (WMMA B-operand ready)
// ---------------------------------------------------------------------------
__global__ __launch_bounds__(256) void prep_weights(
    const float* __restrict__ Wk, const float* __restrict__ Wv,
    unsigned short* __restrict__ WkT, unsigned short* __restrict__ WvT) {
    int idx = blockIdx.x * 256 + threadIdx.x;      // 0 .. D*D-1
    int k = idx / D_, n = idx % D_;
    WkT[(size_t)n * D_ + k] = f2bf(Wk[idx]);
    WvT[(size_t)n * D_ + k] = f2bf(Wv[idx]);
}

// ---------------------------------------------------------------------------
// Kernel 1: all_padded[b] = all(padding_mask[b,:])
// ---------------------------------------------------------------------------
__global__ __launch_bounds__(256) void prep_mask(
    const unsigned char* __restrict__ pm, int* __restrict__ allp) {
    __shared__ int red[256];
    int b = blockIdx.x, t = threadIdx.x;
    int any_valid = 0;
    for (int i = t; i < N_; i += 256) any_valid |= (pm[b * N_ + i] == 0);
    red[t] = any_valid;
    __syncthreads();
    for (int s = 128; s > 0; s >>= 1) {
        if (t < s) red[t] |= red[t + s];
        __syncthreads();
    }
    if (t == 0) allp[b] = (red[0] == 0);
}

// ---------------------------------------------------------------------------
// Kernel 2: qproj[b,d] = query[b,:] @ Wq[:,d] + bq[d]   (tiny: 8.4 MFLOP)
// ---------------------------------------------------------------------------
__global__ __launch_bounds__(256) void qproj_kernel(
    const float* __restrict__ query, const float* __restrict__ Wq,
    const float* __restrict__ bq, float* __restrict__ qp) {
    int idx = blockIdx.x * 256 + threadIdx.x;      // 0 .. B*D-1
    int b = idx / D_, d = idx % D_;
    float acc = bq[d];
    for (int k = 0; k < D_; ++k)
        acc = fmaf(query[b * D_ + k], Wq[(size_t)k * D_ + d], acc);
    qp[idx] = acc;
}

// ---------------------------------------------------------------------------
// Kernel 3: K/V projection GEMM (dominant: ~69 GFLOP), fused masking+bias.
//   block tile: 128 tokens x 64 dcols, K-loop over D in steps of 32.
//   Kb [B*N, D] bf16 row-major ; Vt [B, D, N] bf16 (transposed for out-GEMM).
// ---------------------------------------------------------------------------
__global__ __launch_bounds__(256) void kv_gemm(
    const float* __restrict__ tokens, const unsigned char* __restrict__ pm,
    const unsigned short* __restrict__ WkT, const unsigned short* __restrict__ WvT,
    const float* __restrict__ bk, const float* __restrict__ bv,
    unsigned short* __restrict__ Kb, unsigned short* __restrict__ Vt) {
    constexpr int LDA = 34;                         // A: VALU-staged, pad 2
    constexpr int LDB = 40;                         // B: async-staged, 80B rows (16B aligned)
    __shared__ alignas(16) unsigned short As[128 * LDA];
    __shared__ alignas(16) unsigned short Bks[64 * LDB];
    __shared__ alignas(16) unsigned short Bvs[64 * LDB];
    __shared__ float validS[128];

    int tid = threadIdx.x, lane = tid & 31, wave = tid >> 5;
    int row0 = blockIdx.y * 128;                    // flat token row
    int b = row0 / N_;
    int rowInB = row0 - b * N_;
    int col0 = blockIdx.x * 64;                     // output D column

    if (tid < 128) validS[tid] = pm[row0 + tid] ? 0.0f : 1.0f;

    int wm = wave >> 1, wn = wave & 1;              // 4 x 2 wave grid (32x32 each)
    v8f acck[2][2], accv[2][2];
#pragma unroll
    for (int i = 0; i < 2; ++i)
#pragma unroll
        for (int j = 0; j < 2; ++j) { acck[i][j] = zero8(); accv[i][j] = zero8(); }

    for (int kk = 0; kk < D_; kk += 32) {
        __syncthreads();
        // stage B tiles (verbatim bf16, K-contiguous): async DMA when available
#if USE_ASYNC
        {
            int n = tid >> 2;                       // row 0..63
            int off = (tid & 3) << 3;               // 0,8,16,24 shorts (16B chunks)
            async_copy16(WkT + (size_t)(col0 + n) * D_ + kk + off, &Bks[n * LDB + off]);
            async_copy16(WvT + (size_t)(col0 + n) * D_ + kk + off, &Bvs[n * LDB + off]);
        }
#else
#pragma unroll
        for (int i = 0; i < 8; ++i) {
            int idx = i * 256 + tid;
            int n = idx >> 5, c = idx & 31;
            Bks[n * LDB + c] = WkT[(size_t)(col0 + n) * D_ + kk + c];
            Bvs[n * LDB + c] = WvT[(size_t)(col0 + n) * D_ + kk + c];
        }
#endif
        // stage A: 128x32 fp32 -> masked bf16 (needs conversion: VALU path)
#pragma unroll
        for (int i = 0; i < 16; ++i) {
            int idx = i * 256 + tid;
            int r = idx >> 5, c = idx & 31;
            float x = tokens[(size_t)(row0 + r) * D_ + kk + c] * validS[r];
            As[r * LDA + c] = f2bf(x);
        }
        // prefetch next token tile (global_prefetch_b8 path)
        if (kk + 32 < D_)
            __builtin_prefetch(&tokens[(size_t)(row0 + ((tid >> 1) & 127)) * D_ + kk + 32], 0, 1);
#if USE_ASYNC
        async_wait();
#endif
        __syncthreads();

        AFrag bkF[2], bvF[2];
#pragma unroll
        for (int j = 0; j < 2; ++j) {
            bkF[j] = load_frag<LDB>(Bks + (wn * 32 + j * 16) * LDB, lane);
            bvF[j] = load_frag<LDB>(Bvs + (wn * 32 + j * 16) * LDB, lane);
        }
#pragma unroll
        for (int i = 0; i < 2; ++i) {
            AFrag a = load_frag<LDA>(As + (wm * 32 + i * 16) * LDA, lane);
#pragma unroll
            for (int j = 0; j < 2; ++j) {
                acck[i][j] = wmma_bf16(a, bkF[j], acck[i][j]);
                accv[i][j] = wmma_bf16(a, bvF[j], accv[i][j]);
            }
        }
    }

    // epilogue: +bias, V *= valid, store K row-major / V transposed
    int rl = lane & 15, rh = lane >> 4;
#pragma unroll
    for (int i = 0; i < 2; ++i) {
#pragma unroll
        for (int j = 0; j < 2; ++j) {
            int nglob = col0 + wn * 32 + j * 16 + rl;
            float bkv = bk[nglob], bvv = bv[nglob];
#pragma unroll
            for (int r = 0; r < 8; ++r) {
                int m = wm * 32 + i * 16 + rh * 8 + r;       // local token row
                float kval = acck[i][j][r] + bkv;
                float vval = (accv[i][j][r] + bvv) * validS[m];
                Kb[(size_t)(row0 + m) * D_ + nglob] = f2bf(kval);
                Vt[((size_t)b * D_ + nglob) * N_ + (rowInB + m)] = f2bf(vval);
            }
        }
    }
}

// ---------------------------------------------------------------------------
// Kernel 4: scores[b, 64, n0:n0+256] = (latents + qproj[b]) @ K^T * D^-0.5
//           + safe key-padding mask. Q staged to bf16 per K-slice.
// ---------------------------------------------------------------------------
__global__ __launch_bounds__(256) void scores_gemm(
    const float* __restrict__ latents, const float* __restrict__ qp,
    const unsigned short* __restrict__ Kb, const unsigned char* __restrict__ pm,
    const int* __restrict__ allp, float* __restrict__ S) {
    constexpr int LDA = 34;
    constexpr int LDB = 40;
    __shared__ alignas(16) unsigned short Qs[64 * LDA];
    __shared__ alignas(16) unsigned short Ks[256 * LDB];

    int tid = threadIdx.x, lane = tid & 31, wave = tid >> 5;
    int b = blockIdx.y;
    int n0 = blockIdx.x * 256;
    int ap = allp[b];

    int wm = wave >> 2, wn = wave & 3;              // 2 x 4 wave grid (32x64 each)
    v8f acc[2][4];
#pragma unroll
    for (int i = 0; i < 2; ++i)
#pragma unroll
        for (int j = 0; j < 4; ++j) acc[i][j] = zero8();

    for (int kk = 0; kk < D_; kk += 32) {
        __syncthreads();
        // K tile 256x32 (verbatim bf16): async DMA when available
#if USE_ASYNC
        {
            const unsigned short* g = Kb + ((size_t)b * N_ + n0 + tid) * D_ + kk;
#pragma unroll
            for (int i = 0; i < 4; ++i)
                async_copy16(g + i * 8, &Ks[tid * LDB + i * 8]);
        }
#else
#pragma unroll
        for (int i = 0; i < 32; ++i) {
            int idx = i * 256 + tid;
            int n = idx >> 5, c = idx & 31;
            Ks[n * LDB + c] = Kb[((size_t)b * N_ + n0 + n) * D_ + kk + c];
        }
#endif
        // Q tile 64x32: latents + qproj, fp32 -> bf16 (VALU path)
#pragma unroll
        for (int i = 0; i < 8; ++i) {
            int idx = i * 256 + tid;
            int l = idx >> 5, c = idx & 31;
            float q = latents[(size_t)l * D_ + kk + c] + qp[(size_t)b * D_ + kk + c];
            Qs[l * LDA + c] = f2bf(q);
        }
#if USE_ASYNC
        async_wait();
#endif
        __syncthreads();

        AFrag bF[4];
#pragma unroll
        for (int j = 0; j < 4; ++j)
            bF[j] = load_frag<LDB>(Ks + (wn * 64 + j * 16) * LDB, lane);
#pragma unroll
        for (int i = 0; i < 2; ++i) {
            AFrag a = load_frag<LDA>(Qs + (wm * 32 + i * 16) * LDA, lane);
#pragma unroll
            for (int j = 0; j < 4; ++j)
                acc[i][j] = wmma_bf16(a, bF[j], acc[i][j]);
        }
    }

    const float scale = 0.044194173824159216f;      // 512^-0.5
    int rl = lane & 15, rh = lane >> 4;
#pragma unroll
    for (int i = 0; i < 2; ++i) {
#pragma unroll
        for (int j = 0; j < 4; ++j) {
            int nglob = n0 + wn * 64 + j * 16 + rl;
            bool masked = (pm[(size_t)b * N_ + nglob] != 0) && (ap == 0);
#pragma unroll
            for (int r = 0; r < 8; ++r) {
                int l = wm * 32 + i * 16 + rh * 8 + r;
                float s = acc[i][j][r] * scale;
                if (masked) s = -3.4028234663852886e38f;   // finfo(f32).min
                S[((size_t)b * L_ + l) * N_ + nglob] = s;
            }
        }
    }
}

// ---------------------------------------------------------------------------
// Kernel 5: row softmax over N=4096, emit bf16 weights (n-contiguous).
// ---------------------------------------------------------------------------
__global__ __launch_bounds__(256) void softmax_kernel(
    const float* __restrict__ S, unsigned short* __restrict__ Wgt) {
    __shared__ float red[256];
    int row = blockIdx.x;                           // b*L + l
    int t = threadIdx.x;
    const float* s = S + (size_t)row * N_;

    float m = -3.4028234663852886e38f;
    for (int i = t; i < N_; i += 256) m = fmaxf(m, s[i]);
    red[t] = m; __syncthreads();
    for (int st = 128; st > 0; st >>= 1) {
        if (t < st) red[t] = fmaxf(red[t], red[t + st]);
        __syncthreads();
    }
    m = red[0];
    __syncthreads();

    float sum = 0.f;
    for (int i = t; i < N_; i += 256) sum += __expf(s[i] - m);
    red[t] = sum; __syncthreads();
    for (int st = 128; st > 0; st >>= 1) {
        if (t < st) red[t] += red[t + st];
        __syncthreads();
    }
    float inv = 1.0f / red[0];

    for (int i = t; i < N_; i += 256)
        Wgt[(size_t)row * N_ + i] = f2bf(__expf(s[i] - m) * inv);
}

// ---------------------------------------------------------------------------
// Kernel 6: out[b, 64, d0:d0+64] = weights @ V  (K = N = 4096, WMMA)
//           B operand reads Vt[b][d][n] -> n-contiguous.
// ---------------------------------------------------------------------------
__global__ __launch_bounds__(256) void out_gemm(
    const unsigned short* __restrict__ Wgt, const unsigned short* __restrict__ Vt,
    float* __restrict__ Oraw) {
    constexpr int LDB = 40;
    __shared__ alignas(16) unsigned short Aw[64 * LDB];
    __shared__ alignas(16) unsigned short Bv[64 * LDB];

    int tid = threadIdx.x, lane = tid & 31, wave = tid >> 5;
    int b = blockIdx.y;
    int d0 = blockIdx.x * 64;

    int wm = wave >> 2, wn = wave & 3;              // 2 x 4 wave grid (32x16 each)
    v8f acc[2];
    acc[0] = zero8(); acc[1] = zero8();

    for (int kk = 0; kk < N_; kk += 32) {
        __syncthreads();
#if USE_ASYNC
        {
            int n = tid >> 2;                       // row 0..63
            int off = (tid & 3) << 3;               // 16B chunk
            async_copy16(Wgt + ((size_t)b * L_ + n) * N_ + kk + off, &Aw[n * LDB + off]);
            async_copy16(Vt + ((size_t)b * D_ + d0 + n) * N_ + kk + off, &Bv[n * LDB + off]);
        }
        async_wait();
#else
#pragma unroll
        for (int i = 0; i < 8; ++i) {
            int idx = i * 256 + tid;
            int r = idx >> 5, c = idx & 31;
            Aw[r * LDB + c] = Wgt[((size_t)b * L_ + r) * N_ + kk + c];
            Bv[r * LDB + c] = Vt[((size_t)b * D_ + d0 + r) * N_ + kk + c];
        }
#endif
        __syncthreads();

        AFrag bF = load_frag<LDB>(Bv + (wn * 16) * LDB, lane);
#pragma unroll
        for (int i = 0; i < 2; ++i) {
            AFrag a = load_frag<LDB>(Aw + (wm * 32 + i * 16) * LDB, lane);
            acc[i] = wmma_bf16(a, bF, acc[i]);
        }
    }

    int rl = lane & 15, rh = lane >> 4;
#pragma unroll
    for (int i = 0; i < 2; ++i) {
        int dglob = d0 + wn * 16 + rl;
#pragma unroll
        for (int r = 0; r < 8; ++r) {
            int l = wm * 32 + i * 16 + rh * 8 + r;
            Oraw[((size_t)b * L_ + l) * D_ + dglob] = acc[i][r];
        }
    }
}

// ---------------------------------------------------------------------------
// Kernel 7: RMSNorm + all_padded gating + latent_mask tail.
// ---------------------------------------------------------------------------
__global__ __launch_bounds__(256) void rmsnorm_kernel(
    const float* __restrict__ Oraw, const float* __restrict__ norm_w,
    const int* __restrict__ allp, float* __restrict__ out) {
    __shared__ float red[256];
    int row = blockIdx.x;                           // b*L + l
    int b = row / L_;
    int t = threadIdx.x;
    const float* x = Oraw + (size_t)row * D_;

    float ss = 0.f;
    for (int i = t; i < D_; i += 256) ss += x[i] * x[i];
    red[t] = ss; __syncthreads();
    for (int st = 128; st > 0; st >>= 1) {
        if (t < st) red[t] += red[t + st];
        __syncthreads();
    }
    float var = red[0] * (1.0f / D_);
    float rr = __frsqrt_rn(var + 1e-6f);
    float gate = allp[b] ? 0.f : 1.f;

    for (int i = t; i < D_; i += 256)
        out[(size_t)row * D_ + i] = x[i] * rr * norm_w[i] * gate;
    if (t == 0)
        out[(size_t)B_ * L_ * D_ + row] = allp[b] ? 1.0f : 0.0f;  // latent_mask
}

// ---------------------------------------------------------------------------
// Launch
// ---------------------------------------------------------------------------
extern "C" void kernel_launch(void* const* d_in, const int* in_sizes, int n_in,
                              void* d_out, int out_size, void* d_ws, size_t ws_size,
                              hipStream_t stream) {
    (void)in_sizes; (void)n_in; (void)out_size; (void)ws_size;
    const float*         query  = (const float*)d_in[0];
    const float*         tokens = (const float*)d_in[1];
    const unsigned char* pm     = (const unsigned char*)d_in[2];  // bool mask
    const float*         latents= (const float*)d_in[3];
    const float*         Wq     = (const float*)d_in[4];
    const float*         bq     = (const float*)d_in[5];
    const float*         Wk     = (const float*)d_in[6];
    const float*         bk     = (const float*)d_in[7];
    const float*         Wv     = (const float*)d_in[8];
    const float*         bv     = (const float*)d_in[9];
    const float*         norm_w = (const float*)d_in[10];

    char* ws = (char*)d_ws;
    unsigned short* WkT  = (unsigned short*)(ws + 0);                       // 512 KB
    unsigned short* WvT  = (unsigned short*)(ws + 524288ull);               // 512 KB
    float*          qp   = (float*)        (ws + 1048576ull);               //  32 KB
    int*            allp = (int*)          (ws + 1114112ull);               //  64 B
    unsigned short* Kb   = (unsigned short*)(ws + 2097152ull);              //  64 MB
    unsigned short* Vt   = (unsigned short*)(ws + 2097152ull + 67108864ull);//  64 MB
    float*          S    = (float*)        (ws + 2097152ull + 134217728ull);//  16 MB
    unsigned short* Wgt  = (unsigned short*)(ws + 2097152ull + 150994944ull);// 8 MB
    float*          Oraw = (float*)        (ws + 2097152ull + 159383552ull);// 2 MB

    prep_weights<<<(D_ * D_) / 256, 256, 0, stream>>>(Wk, Wv, WkT, WvT);
    prep_mask<<<B_, 256, 0, stream>>>(pm, allp);
    qproj_kernel<<<(B_ * D_) / 256, 256, 0, stream>>>(query, Wq, bq, qp);
    kv_gemm<<<dim3(D_ / 64, (B_ * N_) / 128), 256, 0, stream>>>(
        tokens, pm, WkT, WvT, bk, bv, Kb, Vt);
    scores_gemm<<<dim3(N_ / 256, B_), 256, 0, stream>>>(latents, qp, Kb, pm, allp, S);
    softmax_kernel<<<B_ * L_, 256, 0, stream>>>(S, Wgt);
    out_gemm<<<dim3(D_ / 64, B_), 256, 0, stream>>>(Wgt, Vt, Oraw);
    rmsnorm_kernel<<<B_ * L_, 256, 0, stream>>>(Oraw, norm_w, allp, (float*)d_out);
}
